// Informer_51393578664229
// MI455X (gfx1250) — matrix operand order repaired and checked by reading, not compiled, not run on hardware
//
#include <hip/hip_runtime.h>
#include <hip/hip_bf16.h>
#include <hip/hip_fp16.h>

typedef _Float16 h16 __attribute__((ext_vector_type(16)));
typedef _Float16 h8  __attribute__((ext_vector_type(8)));
typedef float    f8  __attribute__((ext_vector_type(8)));

#define B_   8
#define L_   2048
#define CIN_ 3
#define D_   256
#define H_   8
#define DH_  32
#define DFF_ 1024
#define EL_  3
#define NC_  10
#define SK_  40      // SAMPLE_K = min(5*ceil(log(2048)), 2048) = 40
#define UT_  40      // U_TOP   = 40
#define M_   (B_ * L_)          // 16384 rows of activations
#define SCALE_ 0.17677669529663687f   // 1/sqrt(32)

// ---------------------------------------------------------------- utilities

__global__ void f32_to_f16_kernel(const float* __restrict__ src,
                                  _Float16* __restrict__ dst, int n) {
  int i = blockIdx.x * 256 + threadIdx.x;
  if (i < n) dst[i] = (_Float16)src[i];
}

// ------------------------------------------------------- token embed + PE

__global__ void embed_kernel(const float* __restrict__ xe,   // (B,L,CIN)
                             const float* __restrict__ pe,   // (L,D)
                             const float* __restrict__ cw,   // (D,CIN,3)
                             float* __restrict__ x,          // (B,L,D)
                             _Float16* __restrict__ xh) {
  int idx = blockIdx.x * 256 + threadIdx.x;
  if (idx >= B_ * L_ * D_) return;
  int d = idx % D_;
  int l = (idx / D_) % L_;
  int b = idx / (D_ * L_);
  float acc = pe[l * D_ + d];
#pragma unroll
  for (int kk = 0; kk < 3; ++kk) {
    int p = l + kk;                       // index into circular pad [0, L+1]
    int sl = (p == 0) ? (L_ - 1) : ((p <= L_) ? (p - 1) : 0);
    const float* xr = xe + ((size_t)b * L_ + sl) * CIN_;
    const float* wr = cw + d * 9 + kk;    // cw[d,c,kk] = cw[d*9 + c*3 + kk]
    acc += xr[0] * wr[0] + xr[1] * wr[3] + xr[2] * wr[6];
  }
  x[idx]  = acc;
  xh[idx] = (_Float16)acc;
}

// --------------------------------------------------------- WMMA GEMM
// C[m,n] = sum_k A[m,k] * W[n,k] + bias[n]
// A: (M,K) f16 row-major, W: (N,K) f16 row-major, C: (M,N) f32.
// One wave per block; wave computes a 16(M) x 64(N) strip.
__global__ __launch_bounds__(32)
void gemm_wmma_kernel(const _Float16* __restrict__ A,
                      const _Float16* __restrict__ W,
                      const float* __restrict__ bias,
                      float* __restrict__ C,
                      int M, int N, int K) {
  const int lane  = threadIdx.x;
  const int m0    = blockIdx.x * 16;
  const int n0    = blockIdx.y * 64;
  const int mrow  = m0 + (lane & 15);
  const int koffA = (lane >> 4) * 16;   // A layout: hi half-wave holds K 16..31
  const int nlane = lane & 15;
  const int kbsB  = (lane >> 4) * 16;   // B layout: hi half-wave holds K 16..31

  f8 acc0 = {}, acc1 = {}, acc2 = {}, acc3 = {};

  const _Float16* arowb = A + (size_t)mrow * K + koffA;
  const _Float16* wrowb = W + (size_t)(n0 + nlane) * K + kbsB;

  for (int kt = 0; kt < K; kt += 32) {
    // A fragment: 16 halves per lane = two contiguous 8-half (16B) loads
    h8 alo = *(const h8*)(arowb + kt);
    h8 ahi = *(const h8*)(arowb + kt + 8);
    h16 a;
#pragma unroll
    for (int i = 0; i < 8; ++i) { a[i] = alo[i]; a[8 + i] = ahi[i]; }

    // B fragments: per lane, 16 contiguous halves of the weight row (32B)
    h16 b0 = *(const h16*)(wrowb + kt);
    h16 b1 = *(const h16*)(wrowb + kt + (size_t)16 * K);
    h16 b2 = *(const h16*)(wrowb + kt + (size_t)32 * K);
    h16 b3 = *(const h16*)(wrowb + kt + (size_t)48 * K);

    acc0 = __builtin_amdgcn_wmma_f32_16x16x32_f16(false, a, false, b0, (short)0, acc0, false, false);
    acc1 = __builtin_amdgcn_wmma_f32_16x16x32_f16(false, a, false, b1, (short)0, acc1, false, false);
    acc2 = __builtin_amdgcn_wmma_f32_16x16x32_f16(false, a, false, b2, (short)0, acc2, false, false);
    acc3 = __builtin_amdgcn_wmma_f32_16x16x32_f16(false, a, false, b3, (short)0, acc3, false, false);
  }

  const int mo = m0 + (lane >> 4) * 8;
  const int no = n0 + nlane;
  const float bb0 = bias[no];
  const float bb1 = bias[no + 16];
  const float bb2 = bias[no + 32];
  const float bb3 = bias[no + 48];
#pragma unroll
  for (int r = 0; r < 8; ++r) {
    float* crow = C + (size_t)(mo + r) * N + no;
    crow[0]  = acc0[r] + bb0;
    crow[16] = acc1[r] + bb1;
    crow[32] = acc2[r] + bb2;
    crow[48] = acc3[r] + bb3;
  }
}

// ----------------------------------------- ProbSparse: sampled-QK measure
// m[b,h,l] = max_s(q_l . k_sample) - sum_s(q_l . k_sample)/L
__global__ void probm_kernel(const float* __restrict__ q,
                             const float* __restrict__ k,
                             const int* __restrict__ isamp,   // (L, SK)
                             float* __restrict__ m) {
  int t = blockIdx.x * 256 + threadIdx.x;
  if (t >= B_ * H_ * L_) return;
  int l = t % L_;
  int h = (t / L_) % H_;
  int b = t / (L_ * H_);
  const float* qr = q + ((size_t)b * L_ + l) * D_ + h * DH_;
  float qv[DH_];
#pragma unroll
  for (int d = 0; d < DH_; ++d) qv[d] = qr[d];
  float mx = -3.4e38f, sm = 0.f;
  for (int s = 0; s < SK_; ++s) {
    int ks = isamp[l * SK_ + s];
    const float* kr = k + ((size_t)b * L_ + ks) * D_ + h * DH_;
    float dot = 0.f;
#pragma unroll
    for (int d = 0; d < DH_; ++d) dot += qv[d] * kr[d];
    mx = fmaxf(mx, dot);
    sm += dot;
  }
  m[t] = mx - sm * (1.0f / (float)L_);
}

// ------------------------------------------------- top-40 queries per (b,h)
__global__ __launch_bounds__(256)
void topk_kernel(const float* __restrict__ m, int* __restrict__ topidx) {
  __shared__ float sv[L_];
  __shared__ float rmax[256];
  __shared__ int   ridx[256];
  int bh = blockIdx.x;
  int tid = threadIdx.x;
  const float* mr = m + (size_t)bh * L_;
  for (int i = tid; i < L_; i += 256) sv[i] = mr[i];
  __syncthreads();
  for (int u = 0; u < UT_; ++u) {
    float best = -3.4e38f; int bi = 0;
    for (int i = tid; i < L_; i += 256) {
      float vv = sv[i];
      if (vv > best) { best = vv; bi = i; }
    }
    rmax[tid] = best; ridx[tid] = bi;
    __syncthreads();
    for (int st = 128; st > 0; st >>= 1) {
      if (tid < st) {
        if (rmax[tid + st] > rmax[tid] ||
            (rmax[tid + st] == rmax[tid] && ridx[tid + st] < ridx[tid])) {
          rmax[tid] = rmax[tid + st]; ridx[tid] = ridx[tid + st];
        }
      }
      __syncthreads();
    }
    if (tid == 0) {
      topidx[bh * UT_ + u] = ridx[0];
      sv[ridx[0]] = -3.4e38f;
    }
    __syncthreads();
  }
}

// ----------------------------------------------------------- mean of V
__global__ void vmean_kernel(const float* __restrict__ v, float* __restrict__ vm) {
  int t = blockIdx.x * 256 + threadIdx.x;   // B*H*DH
  if (t >= B_ * H_ * DH_) return;
  int d = t % DH_;
  int h = (t / DH_) % H_;
  int b = t / (DH_ * H_);
  float s = 0.f;
  for (int l = 0; l < L_; ++l)
    s += v[((size_t)b * L_ + l) * D_ + h * DH_ + d];
  vm[t] = s * (1.0f / (float)L_);
}

// --------------------------- full attention for the 40 selected queries
__global__ __launch_bounds__(256)
void attn_kernel(const float* __restrict__ q, const float* __restrict__ k,
                 const float* __restrict__ v, const int* __restrict__ topidx,
                 float* __restrict__ upd) {
  __shared__ float sc[L_];
  __shared__ float qs[DH_];
  __shared__ float red[256];
  __shared__ float part[8][DH_];
  int blk = blockIdx.x;              // bh*UT + u
  int u  = blk % UT_;
  int bh = blk / UT_;
  int h  = bh % H_;
  int b  = bh / H_;
  int tid = threadIdx.x;
  int lq = topidx[bh * UT_ + u];
  if (tid < DH_) qs[tid] = q[((size_t)b * L_ + lq) * D_ + h * DH_ + tid];
  __syncthreads();

  float lmax = -3.4e38f;
  for (int key = tid; key < L_; key += 256) {
    const float* kr = k + ((size_t)b * L_ + key) * D_ + h * DH_;
    float dot = 0.f;
#pragma unroll
    for (int d = 0; d < DH_; ++d) dot += qs[d] * kr[d];
    dot *= SCALE_;
    sc[key] = dot;
    lmax = fmaxf(lmax, dot);
  }
  red[tid] = lmax; __syncthreads();
  for (int st = 128; st > 0; st >>= 1) {
    if (tid < st) red[tid] = fmaxf(red[tid], red[tid + st]);
    __syncthreads();
  }
  float gmax = red[0];
  __syncthreads();

  float lsum = 0.f;
  for (int key = tid; key < L_; key += 256) {
    float e = expf(sc[key] - gmax);
    sc[key] = e;
    lsum += e;
  }
  red[tid] = lsum; __syncthreads();
  for (int st = 128; st > 0; st >>= 1) {
    if (tid < st) red[tid] += red[tid + st];
    __syncthreads();
  }
  float inv = 1.0f / red[0];

  int g = tid >> 5, d = tid & 31;    // 8 key-chunks x 32 head-dims
  float acc = 0.f;
  for (int key = g * 256; key < (g + 1) * 256; ++key)
    acc += sc[key] * v[((size_t)b * L_ + key) * D_ + h * DH_ + d];
  part[g][d] = acc;
  __syncthreads();
  if (tid < DH_) {
    float s = 0.f;
#pragma unroll
    for (int gg = 0; gg < 8; ++gg) s += part[gg][tid];
    upd[((size_t)bh * UT_ + u) * DH_ + tid] = s * inv;
  }
}

// ----------------------------- context: broadcast mean, scatter updates
__global__ void ctxfill_kernel(const float* __restrict__ vm, float* __restrict__ ctx) {
  int idx = blockIdx.x * 256 + threadIdx.x;
  if (idx >= B_ * L_ * D_) return;
  int d = idx % D_;
  int b = idx / (D_ * L_);
  int h = d / DH_, dd = d % DH_;
  ctx[idx] = vm[(b * H_ + h) * DH_ + dd];
}

__global__ void ctxscatter_kernel(const float* __restrict__ upd,
                                  const int* __restrict__ topidx,
                                  float* __restrict__ ctx) {
  int t = blockIdx.x * 256 + threadIdx.x;   // B*H*UT*DH
  if (t >= B_ * H_ * UT_ * DH_) return;
  int d = t % DH_;
  int u = (t / DH_) % UT_;
  int h = (t / (DH_ * UT_)) % H_;
  int b = t / (DH_ * UT_ * H_);
  int l = topidx[(b * H_ + h) * UT_ + u];
  ctx[((size_t)b * L_ + l) * D_ + h * DH_ + d] = upd[t];
}

// ---------------------------------------------- residual add + LayerNorm
__global__ __launch_bounds__(256)
void add_ln_kernel(const float* __restrict__ xin, const float* __restrict__ res,
                   const float* __restrict__ g, const float* __restrict__ bta,
                   float* __restrict__ xout, _Float16* __restrict__ xh) {
  __shared__ float red[256];
  int row = blockIdx.x;
  int tid = threadIdx.x;
  size_t o = (size_t)row * D_ + tid;
  float val = xin[o] + res[o];
  red[tid] = val; __syncthreads();
  for (int st = 128; st > 0; st >>= 1) {
    if (tid < st) red[tid] += red[tid + st];
    __syncthreads();
  }
  float mean = red[0] * (1.0f / D_);
  __syncthreads();
  float dv = val - mean;
  red[tid] = dv * dv; __syncthreads();
  for (int st = 128; st > 0; st >>= 1) {
    if (tid < st) red[tid] += red[tid + st];
    __syncthreads();
  }
  float var = red[0] * (1.0f / D_);
  float nv = dv * rsqrtf(var + 1e-5f) * g[tid] + bta[tid];
  xout[o] = nv;
  xh[o]   = (_Float16)nv;
}

// -------------------------------------------------- exact GELU -> f16
__global__ void gelu_kernel(const float* __restrict__ y,
                            _Float16* __restrict__ yh, int n) {
  int i = blockIdx.x * 256 + threadIdx.x;
  if (i >= n) return;
  float xv = y[i];
  float gv = 0.5f * xv * (1.0f + erff(xv * 0.7071067811865475f));
  yh[i] = (_Float16)gv;
}

// --------------------------------------- final LayerNorm * time-mark mask
__global__ __launch_bounds__(256)
void final_ln_kernel(const float* __restrict__ x, const float* __restrict__ g,
                     const float* __restrict__ bta, const float* __restrict__ mark,
                     float* __restrict__ xf) {
  __shared__ float red[256];
  int row = blockIdx.x;
  int tid = threadIdx.x;
  size_t o = (size_t)row * D_ + tid;
  float val = x[o];
  red[tid] = val; __syncthreads();
  for (int st = 128; st > 0; st >>= 1) {
    if (tid < st) red[tid] += red[tid + st];
    __syncthreads();
  }
  float mean = red[0] * (1.0f / D_);
  __syncthreads();
  float dv = val - mean;
  red[tid] = dv * dv; __syncthreads();
  for (int st = 128; st > 0; st >>= 1) {
    if (tid < st) red[tid] += red[tid + st];
    __syncthreads();
  }
  float var = red[0] * (1.0f / D_);
  float nv = dv * rsqrtf(var + 1e-5f) * g[tid] + bta[tid];
  xf[o] = nv * mark[row];
}

// ------------------------------------------------------------- final FC
__global__ __launch_bounds__(256)
void fc_kernel(const float* __restrict__ xf, const float* __restrict__ fw,
               const float* __restrict__ fb, float* __restrict__ out) {
  __shared__ float red[256];
  int bc = blockIdx.x;                 // b*NC + c
  int c = bc % NC_;
  int b = bc / NC_;
  int tid = threadIdx.x;
  const float* xr = xf + (size_t)b * L_ * D_;
  const float* wr = fw + (size_t)c * L_ * D_;
  float s = 0.f;
  for (int i = tid; i < L_ * D_; i += 256) s += xr[i] * wr[i];
  red[tid] = s; __syncthreads();
  for (int st = 128; st > 0; st >>= 1) {
    if (tid < st) red[tid] += red[tid + st];
    __syncthreads();
  }
  if (tid == 0) out[bc] = red[0] + fb[c];
}

// ================================================================ launch

extern "C" void kernel_launch(void* const* d_in, const int* in_sizes, int n_in,
                              void* d_out, int out_size, void* d_ws, size_t ws_size,
                              hipStream_t stream) {
  (void)in_sizes; (void)n_in; (void)out_size; (void)ws_size;

  const float* x_enc  = (const float*)d_in[0];
  const float* mark   = (const float*)d_in[1];
  const int*   isamp  = (const int*)  d_in[2];
  const float* pe     = (const float*)d_in[3];
  const float* conv_w = (const float*)d_in[4];
  const float* Wq = (const float*)d_in[5];
  const float* bq = (const float*)d_in[6];
  const float* Wk = (const float*)d_in[7];
  const float* bk = (const float*)d_in[8];
  const float* Wv = (const float*)d_in[9];
  const float* bv = (const float*)d_in[10];
  const float* Wo = (const float*)d_in[11];
  const float* bo = (const float*)d_in[12];
  const float* c1w = (const float*)d_in[13];
  const float* c1b = (const float*)d_in[14];
  const float* c2w = (const float*)d_in[15];
  const float* c2b = (const float*)d_in[16];
  const float* n1g = (const float*)d_in[17];
  const float* n1b = (const float*)d_in[18];
  const float* n2g = (const float*)d_in[19];
  const float* n2b = (const float*)d_in[20];
  const float* ng  = (const float*)d_in[21];
  const float* nb  = (const float*)d_in[22];
  const float* fcw = (const float*)d_in[23];
  const float* fcb = (const float*)d_in[24];
  float* out = (float*)d_out;

  // ---- workspace carve-out (256B aligned)
  char* ws = (char*)d_ws;
  size_t off = 0;
  auto carve = [&](size_t bytes) -> void* {
    void* p = ws + off;
    off += (bytes + 255) & ~(size_t)255;
    return p;
  };
  const size_t BLD = (size_t)B_ * L_ * D_;       // 4,194,304
  const size_t BLF = (size_t)B_ * L_ * DFF_;     // 16,777,216

  float*    x     = (float*)   carve(BLD * 4);
  _Float16* x_h   = (_Float16*)carve(BLD * 2);
  float*    q     = (float*)   carve(BLD * 4);   // q,k,v,ctx contiguous:
  float*    kbuf  = (float*)   carve(BLD * 4);   //   aliased by y (B,L,DFF)
  float*    vbuf  = (float*)   carve(BLD * 4);   //   after attention is done
  float*    ctx   = (float*)   carve(BLD * 4);
  float*    y     = q;                           // 64 MB alias
  _Float16* ctx_h = (_Float16*)carve(BLD * 2);
  float*    att   = (float*)   carve(BLD * 4);   // Wo out, then c2 out
  _Float16* y_h   = (_Float16*)carve(BLF * 2);
  float*    mv    = (float*)   carve((size_t)B_ * H_ * L_ * 4);
  int*      tix   = (int*)     carve((size_t)B_ * H_ * UT_ * 4);
  float*    upd   = (float*)   carve((size_t)B_ * H_ * UT_ * DH_ * 4);
  float*    vm    = (float*)   carve((size_t)B_ * H_ * DH_ * 4);
  _Float16* Wq_h  = (_Float16*)carve((size_t)EL_ * D_ * D_ * 2);
  _Float16* Wk_h  = (_Float16*)carve((size_t)EL_ * D_ * D_ * 2);
  _Float16* Wv_h  = (_Float16*)carve((size_t)EL_ * D_ * D_ * 2);
  _Float16* Wo_h  = (_Float16*)carve((size_t)EL_ * D_ * D_ * 2);
  _Float16* c1w_h = (_Float16*)carve((size_t)EL_ * DFF_ * D_ * 2);
  _Float16* c2w_h = (_Float16*)carve((size_t)EL_ * D_ * DFF_ * 2);
  float*    xf    = (float*)   carve(BLD * 4);

  auto cvt = [&](const float* s, _Float16* d, int n) {
    f32_to_f16_kernel<<<(n + 255) / 256, 256, 0, stream>>>(s, d, n);
  };

  // ---- weights fp32 -> f16 (every call; no cross-call caching allowed)
  cvt(Wq,  Wq_h,  EL_ * D_ * D_);
  cvt(Wk,  Wk_h,  EL_ * D_ * D_);
  cvt(Wv,  Wv_h,  EL_ * D_ * D_);
  cvt(Wo,  Wo_h,  EL_ * D_ * D_);
  cvt(c1w, c1w_h, EL_ * DFF_ * D_);
  cvt(c2w, c2w_h, EL_ * D_ * DFF_);

  // ---- embedding
  embed_kernel<<<(int)((BLD + 255) / 256), 256, 0, stream>>>(x_enc, pe, conv_w, x, x_h);

  dim3 gD(M_ / 16, D_ / 64);     // N = 256
  dim3 gF(M_ / 16, DFF_ / 64);   // N = 1024

  for (int i = 0; i < EL_; ++i) {
    const _Float16* wq = Wq_h + (size_t)i * D_ * D_;
    const _Float16* wk = Wk_h + (size_t)i * D_ * D_;
    const _Float16* wv = Wv_h + (size_t)i * D_ * D_;
    const _Float16* wo = Wo_h + (size_t)i * D_ * D_;
    const _Float16* w1 = c1w_h + (size_t)i * DFF_ * D_;
    const _Float16* w2 = c2w_h + (size_t)i * D_ * DFF_;

    // Q/K/V projections (WMMA)
    gemm_wmma_kernel<<<gD, 32, 0, stream>>>(x_h, wq, bq + i * D_, q,    M_, D_, D_);
    gemm_wmma_kernel<<<gD, 32, 0, stream>>>(x_h, wk, bk + i * D_, kbuf, M_, D_, D_);
    gemm_wmma_kernel<<<gD, 32, 0, stream>>>(x_h, wv, bv + i * D_, vbuf, M_, D_, D_);

    // ProbSparse attention (fp32, tiny FLOPs)
    probm_kernel<<<(B_ * H_ * L_ + 255) / 256, 256, 0, stream>>>(q, kbuf, isamp, mv);
    topk_kernel<<<B_ * H_, 256, 0, stream>>>(mv, tix);
    vmean_kernel<<<(B_ * H_ * DH_ + 255) / 256, 256, 0, stream>>>(vbuf, vm);
    attn_kernel<<<B_ * H_ * UT_, 256, 0, stream>>>(q, kbuf, vbuf, tix, upd);
    ctxfill_kernel<<<(int)((BLD + 255) / 256), 256, 0, stream>>>(vm, ctx);
    ctxscatter_kernel<<<(B_ * H_ * UT_ * DH_ + 255) / 256, 256, 0, stream>>>(upd, tix, ctx);

    // output projection + residual + LN1
    cvt(ctx, ctx_h, (int)BLD);
    gemm_wmma_kernel<<<gD, 32, 0, stream>>>(ctx_h, wo, bo + i * D_, att, M_, D_, D_);
    add_ln_kernel<<<M_, 256, 0, stream>>>(x, att, n1g + i * D_, n1b + i * D_, x, x_h);

    // FFN: gelu(x @ c1w.T + b1) @ c2w.T + b2, residual + LN2
    gemm_wmma_kernel<<<gF, 32, 0, stream>>>(x_h, w1, c1b + i * DFF_, y, M_, DFF_, D_);
    gelu_kernel<<<(int)((BLF + 255) / 256), 256, 0, stream>>>(y, y_h, (int)BLF);
    gemm_wmma_kernel<<<gD, 32, 0, stream>>>(y_h, w2, c2b + i * D_, att, M_, D_, DFF_);
    add_ln_kernel<<<M_, 256, 0, stream>>>(x, att, n2g + i * D_, n2b + i * D_, x, x_h);
  }

  // final LN, mask, classifier head
  final_ln_kernel<<<M_, 256, 0, stream>>>(x, ng, nb, mark, xf);
  fc_kernel<<<B_ * NC_, 256, 0, stream>>>(xf, fcw, fcb, out);
}
